// SpanContextualInteraction_46935402610850
// MI455X (gfx1250) — matrix operand order repaired and checked by reading, not compiled, not run on hardware
//
#include <hip/hip_runtime.h>

typedef __bf16 bf16;
typedef __attribute__((ext_vector_type(16))) __bf16 v16bf;
typedef __attribute__((ext_vector_type(8)))  float  v8f;

#define KB  16      // batch
#define KT  2048    // tokens
#define KH  768     // hidden
#define KHH 384     // HH
#define KS  24      // spans per batch
#define KNH 8       // heads
#define KHD 96      // head dim
#define KW  16      // WMAX
#define KNS (KB*KS) // 384 spans per encoder
#define GKT 64      // GEMM K-tile (double-buffered async staging)

// ---------------------------------------------------------------------------
// WMMA fragment helpers (wave32; contiguous k order, k = 16*(lane/16) + element, for A and B)
// ---------------------------------------------------------------------------
__device__ __forceinline__ float sigmoidf_(float x){ return 1.0f/(1.0f + __expf(-x)); }

// A-matrix 16x32 bf16, row-major source (lda elements per row).
__device__ __forceinline__ v16bf load_a_frag(const bf16* A, int lda, int lane){
  int m = lane & 15, hi = lane >> 4;
  const bf16* row = A + m*lda + 16*hi;
  v16bf a;
#pragma unroll
  for (int r = 0; r < 8; ++r){
    int k = 2*r;
    a[2*r]   = row[k];
    a[2*r+1] = row[k+1];
  }
  return a;
}

// B-matrix 32x16 bf16 from row-major KxN source: element (k,n) = Bsrc[k*ldb + n]
__device__ __forceinline__ v16bf load_b_frag(const bf16* Bsrc, int ldb, int lane){
  int n = lane & 15, hi = lane >> 4;
  const bf16* col = Bsrc + n + (size_t)(16*hi)*ldb;
  v16bf b;
#pragma unroll
  for (int r = 0; r < 8; ++r){
    b[2*r]   = col[(size_t)(2*r)*ldb];
    b[2*r+1] = col[(size_t)(2*r+1)*ldb];
  }
  return b;
}

// B-matrix 32x16 bf16 from a row-major (Nout x K) weight: B(k,n) = W[n*ldw + k]
__device__ __forceinline__ v16bf load_bT_frag(const bf16* W, int ldw, int lane){
  int n = lane & 15, hi = lane >> 4;
  const bf16* row = W + (size_t)n*ldw + 16*hi;
  v16bf b;
#pragma unroll
  for (int r = 0; r < 8; ++r){
    b[2*r]   = row[2*r];
    b[2*r+1] = row[2*r+1];
  }
  return b;
}

__device__ __forceinline__ v8f wmma_bf16(v16bf a, v16bf b, v8f c){
  return __builtin_amdgcn_wmma_f32_16x16x32_bf16(false, a, false, b, (short)0, c, false, false);
}

// Async global -> LDS copy of 64 contiguous bytes (4 x B128), ASYNCcnt-tracked.
__device__ __forceinline__ void async_copy64(bf16* lds_dst, const bf16* gsrc){
  unsigned l0 = (unsigned)(uintptr_t)(void*)lds_dst;  // low 32 bits = LDS offset
#pragma unroll
  for (int q = 0; q < 4; ++q){
    asm volatile("global_load_async_to_lds_b128 %0, %1, off"
                 :: "v"(l0 + 16u*q), "v"(gsrc + 8*q) : "memory");
  }
}

// ---------------------------------------------------------------------------
// fp32 -> bf16 convert
// ---------------------------------------------------------------------------
__global__ void cvt_f32_to_bf16(const float* __restrict__ src, bf16* __restrict__ dst, int n){
  int i = blockIdx.x * blockDim.x + threadIdx.x;
  if (i < n) dst[i] = (bf16)src[i];
}

// ---------------------------------------------------------------------------
// bulk copy hidden_states -> out (float4)
// ---------------------------------------------------------------------------
__global__ void copy_f32x4(const float4* __restrict__ src, float4* __restrict__ dst, int n4){
  int i = blockIdx.x * blockDim.x + threadIdx.x;
  if (i < n4) dst[i] = src[i];
}

// ---------------------------------------------------------------------------
// Gather span windows: Xg[span][16][768] bf16 and x_last rows Xb[span][768].
// ---------------------------------------------------------------------------
__global__ void lstm_gather_kernel(const float* __restrict__ hs, const int* __restrict__ spans,
                                   bf16* __restrict__ Xg, bf16* __restrict__ Xb){
  const int g = blockIdx.x;
  const int b = g / KS, s = g % KS;
  int st = spans[(b*KS + s)*2 + 0];
  int en = spans[(b*KS + s)*2 + 1];
  st = min(max(st, 0), KT - 1);
  en = min(max(en, st + 1), KT);
  const int len = min(en - st, KW);

  for (int i = threadIdx.x; i < KW*KH; i += blockDim.x){
    int t = i / KH, k = i - t*KH;
    int row = min(st + t, KT - 1);
    Xg[(size_t)g*KW*KH + i] = (bf16)hs[((size_t)b*KT + row)*KH + k];
  }
  const int rowL = min(st + len - 1, KT - 1);
  for (int k = threadIdx.x; k < KH; k += blockDim.x)
    Xb[(size_t)g*KH + k] = (bf16)hs[((size_t)b*KT + rowL)*KH + k];
}

// ---------------------------------------------------------------------------
// Blocked GEMM: C(MxN,f32) = A(MxK,bf16 row-major) @ W(NxK,bf16 row-major)^T.
// Block = 128 rows x 64 cols.  A K-tiles (128x64) double-buffered in LDS via
// GLOBAL_LOAD_ASYNC_TO_LDS_B128; B-frag reused over 4 M-tiles.
// M multiple of 128, N multiple of 64, K multiple of 64.
// ---------------------------------------------------------------------------
__global__ void gemm_a_wt_kernel(const bf16* __restrict__ A, const bf16* __restrict__ W,
                                 float* __restrict__ C, int M, int N, int K){
  extern __shared__ char smem[];
  bf16* As0 = (bf16*)smem;                      // [128][64]
  bf16* As1 = As0 + 128*GKT;                    // [128][64]
  const int tid = threadIdx.x, wave = tid >> 5, lane = tid & 31;
  const int m0 = blockIdx.y * 128;
  const int n0 = blockIdx.x * 64 + (wave & 3) * 16;
  const int mg = (wave >> 2) * 4;               // 4 M-tiles per wave

  const int row = tid >> 1, seg = tid & 1;      // 64B per thread per tile
  const bf16* gsrc = A + (size_t)(m0 + row)*K + seg*32;
  bf16* l0 = As0 + row*GKT + seg*32;
  bf16* l1 = As1 + row*GKT + seg*32;

  v8f acc[4] = {{}, {}, {}, {}};
  const int ntiles = K / GKT;

  async_copy64(l0, gsrc);                       // prologue: tile 0 -> As0
  for (int i = 0; i < ntiles; ++i){
    const bf16* cur = (i & 1) ? As1 : As0;
    if (i + 1 < ntiles){
      async_copy64((i & 1) ? l0 : l1, gsrc + (size_t)(i + 1)*GKT);
      asm volatile("s_wait_asynccnt 0x4" ::: "memory");  // tile i complete
    } else {
      asm volatile("s_wait_asynccnt 0x0" ::: "memory");
    }
    __syncthreads();
#pragma unroll
    for (int sub = 0; sub < 2; ++sub){
      int kk = i*GKT + sub*32;
      __builtin_prefetch(W + (size_t)n0*K + kk + GKT, 0, 1);
      v16bf bb = load_bT_frag(W + (size_t)n0*K + kk, K, lane);
#pragma unroll
      for (int mt = 0; mt < 4; ++mt){
        v16bf a = load_a_frag(cur + (mg + mt)*16*GKT + sub*32, GKT, lane);
        acc[mt] = wmma_bf16(a, bb, acc[mt]);
      }
    }
    __syncthreads();   // reads of `cur` done before it is re-targeted
  }
  int n = lane & 15, hi = lane >> 4;
#pragma unroll
  for (int mt = 0; mt < 4; ++mt){
#pragma unroll
    for (int r = 0; r < 8; ++r){
      int m = (mg + mt)*16 + r + 8*hi;
      C[(size_t)(m0 + m)*N + n0 + n] = acc[mt][r];
    }
  }
}

// ---------------------------------------------------------------------------
// LSTM recurrence (input projection precomputed in Gx):
//   G_t = Gx[:,t,:] + H @ Whh^T + b ; gate math; masked state update.
// Grid = 48 blocks: 0..23 aspect encoder, 24..47 opinion encoder.
// ---------------------------------------------------------------------------
__global__ void lstm_rec_kernel(const int* __restrict__ spans_a, const int* __restrict__ spans_o,
                                const float* __restrict__ Gx_a, const float* __restrict__ Gx_o,
                                const bf16* __restrict__ whh_a, const bf16* __restrict__ whh_o,
                                const float* __restrict__ bf_a, const float* __restrict__ bf_o,
                                float* __restrict__ ea, float* __restrict__ eo){
  extern __shared__ char smem[];
  bf16*  hbf  = (bf16*)smem;                 // [16][384]
  float* cst  = (float*)(hbf + 16*KHH);      // [16][384]
  int*   lenS = (int*)(cst + 16*KHH);        // [16]

  const bool isA = blockIdx.x < 24;
  const int  blk = isA ? blockIdx.x : blockIdx.x - 24;
  const int*   spans = isA ? spans_a : spans_o;
  const float* Gx    = isA ? Gx_a  : Gx_o;
  const bf16*  whh   = isA ? whh_a : whh_o;
  const float* bfv   = isA ? bf_a  : bf_o;
  float*       out   = isA ? ea    : eo;

  const int tid = threadIdx.x, wave = tid >> 5, lane = tid & 31;
  const int base = blk * 16;

  if (tid < 16){
    int g  = base + tid;
    int b  = g / KS, s = g % KS;
    int st = spans[(b*KS + s)*2 + 0];
    int en = spans[(b*KS + s)*2 + 1];
    st = min(max(st, 0), KT - 1);
    en = min(max(en, st + 1), KT);
    lenS[tid] = min(en - st, KW);
  }
  for (int i = tid; i < 16*KHH; i += blockDim.x){ hbf[i] = (bf16)0.0f; cst[i] = 0.0f; }
  __syncthreads();

  const int j0 = wave * 48;

  for (int t = 0; t < KW; ++t){
    float hnewA[3][8], cnewA[3][8];
#pragma unroll
    for (int jt = 0; jt < 3; ++jt){
      int jc = j0 + jt*16;
      v8f acc0 = {}, acc1 = {}, acc2 = {}, acc3 = {};
      for (int kk = 0; kk < KHH; kk += 32){
        v16bf a = load_a_frag(hbf + kk, KHH, lane);
        acc0 = wmma_bf16(a, load_bT_frag(whh + (size_t)(0*KHH + jc)*KHH + kk, KHH, lane), acc0);
        acc1 = wmma_bf16(a, load_bT_frag(whh + (size_t)(1*KHH + jc)*KHH + kk, KHH, lane), acc1);
        acc2 = wmma_bf16(a, load_bT_frag(whh + (size_t)(2*KHH + jc)*KHH + kk, KHH, lane), acc2);
        acc3 = wmma_bf16(a, load_bT_frag(whh + (size_t)(3*KHH + jc)*KHH + kk, KHH, lane), acc3);
      }
      int n = lane & 15, hi = lane >> 4;
      int j = jc + n;
#pragma unroll
      for (int r = 0; r < 8; ++r){
        int m = r + 8*hi;
        const float* gx = Gx + ((size_t)(base + m)*KW + t)*(4*KHH);
        float iv = acc0[r] + gx[j]          + bfv[j];
        float fv = acc1[r] + gx[KHH   + j]  + bfv[KHH   + j];
        float gv = acc2[r] + gx[2*KHH + j]  + bfv[2*KHH + j];
        float ov = acc3[r] + gx[3*KHH + j]  + bfv[3*KHH + j];
        float c_old = cst[m*KHH + j];
        float c_new = sigmoidf_(fv)*c_old + sigmoidf_(iv)*tanhf(gv);
        float h_new = sigmoidf_(ov)*tanhf(c_new);
        cnewA[jt][r] = c_new;
        hnewA[jt][r] = h_new;
      }
    }
    __syncthreads();
#pragma unroll
    for (int jt = 0; jt < 3; ++jt){
      int n = lane & 15, hi = lane >> 4;
      int j = j0 + jt*16 + n;
#pragma unroll
      for (int r = 0; r < 8; ++r){
        int m = r + 8*hi;
        if (t < lenS[m]){
          cst[m*KHH + j] = cnewA[jt][r];
          hbf[m*KHH + j] = (bf16)hnewA[jt][r];
        }
      }
    }
    __syncthreads();
  }

  for (int i = tid; i < 16*KHH; i += blockDim.x){
    int m = i / KHH, j = i - m*KHH;
    out[(size_t)(base + m)*KH + j] = (float)hbf[i];
  }
}

// ---------------------------------------------------------------------------
// Backward LSTM cell (elementwise on precomputed Gb = x_last @ Wih_b^T)
// ---------------------------------------------------------------------------
__global__ void lstm_bwd_kernel(const float* __restrict__ Gb, const float* __restrict__ b_b,
                                float* __restrict__ out){
  int i = blockIdx.x * blockDim.x + threadIdx.x;
  if (i >= KNS*KHH) return;
  int g = i / KHH, j = i - g*KHH;
  const float* gb = Gb + (size_t)g*(4*KHH);
  float iv = gb[j]          + b_b[j];
  float gv = gb[2*KHH + j]  + b_b[2*KHH + j];
  float ov = gb[3*KHH + j]  + b_b[3*KHH + j];
  float cb = sigmoidf_(iv)*tanhf(gv);
  out[(size_t)g*KH + KHH + j] = sigmoidf_(ov)*tanhf(cb);
}

// ---------------------------------------------------------------------------
// FALLBACK: fused span bi-LSTM (used only if workspace is too small).
// ---------------------------------------------------------------------------
__global__ void lstm_span_kernel(const float* __restrict__ hs, const int* __restrict__ spans,
                                 const bf16* __restrict__ wih_f, const bf16* __restrict__ whh_f,
                                 const float* __restrict__ b_f,
                                 const bf16* __restrict__ wih_b, const float* __restrict__ b_b,
                                 float* __restrict__ out){
  extern __shared__ char smem[];
  bf16*  xs   = (bf16*)smem;
  bf16*  hbf  = xs + 16*KH;
  float* cst  = (float*)(hbf + 16*KHH);
  int*   meta = (int*)(cst + 16*KHH);

  const int tid  = threadIdx.x;
  const int wave = tid >> 5, lane = tid & 31;
  const int base = blockIdx.x * 16;

  if (tid < 16){
    int g  = base + tid;
    int b  = g / KS, s = g % KS;
    int st = spans[(b*KS + s)*2 + 0];
    int en = spans[(b*KS + s)*2 + 1];
    st = min(max(st, 0), KT - 1);
    en = min(max(en, st + 1), KT);
    meta[tid] = st; meta[16 + tid] = min(en - st, KW); meta[32 + tid] = b;
  }
  for (int i = tid; i < 16*KHH; i += blockDim.x){ hbf[i] = (bf16)0.0f; cst[i] = 0.0f; }
  __syncthreads();

  const int j0 = wave * 48;

  for (int t = 0; t < KW; ++t){
    for (int i = tid; i < 16*KH; i += blockDim.x){
      int m = i / KH, k = i - m*KH;
      int row = min(meta[m] + t, KT - 1);
      xs[i] = (bf16)hs[((size_t)meta[32 + m]*KT + row)*KH + k];
    }
    __syncthreads();

    float hnewA[3][8], cnewA[3][8];
#pragma unroll
    for (int jt = 0; jt < 3; ++jt){
      int jc = j0 + jt*16;
      v8f acc0 = {}, acc1 = {}, acc2 = {}, acc3 = {};
      for (int kk = 0; kk < KH; kk += 32){
        v16bf a = load_a_frag(xs + kk, KH, lane);
        acc0 = wmma_bf16(a, load_bT_frag(wih_f + (size_t)(0*KHH + jc)*KH + kk, KH, lane), acc0);
        acc1 = wmma_bf16(a, load_bT_frag(wih_f + (size_t)(1*KHH + jc)*KH + kk, KH, lane), acc1);
        acc2 = wmma_bf16(a, load_bT_frag(wih_f + (size_t)(2*KHH + jc)*KH + kk, KH, lane), acc2);
        acc3 = wmma_bf16(a, load_bT_frag(wih_f + (size_t)(3*KHH + jc)*KH + kk, KH, lane), acc3);
      }
      for (int kk = 0; kk < KHH; kk += 32){
        v16bf a = load_a_frag(hbf + kk, KHH, lane);
        acc0 = wmma_bf16(a, load_bT_frag(whh_f + (size_t)(0*KHH + jc)*KHH + kk, KHH, lane), acc0);
        acc1 = wmma_bf16(a, load_bT_frag(whh_f + (size_t)(1*KHH + jc)*KHH + kk, KHH, lane), acc1);
        acc2 = wmma_bf16(a, load_bT_frag(whh_f + (size_t)(2*KHH + jc)*KHH + kk, KHH, lane), acc2);
        acc3 = wmma_bf16(a, load_bT_frag(whh_f + (size_t)(3*KHH + jc)*KHH + kk, KHH, lane), acc3);
      }
      int n = lane & 15, hi = lane >> 4;
      int j = jc + n;
#pragma unroll
      for (int r = 0; r < 8; ++r){
        int m = r + 8*hi;
        float iv = acc0[r] + b_f[j];
        float fv = acc1[r] + b_f[KHH   + j];
        float gv = acc2[r] + b_f[2*KHH + j];
        float ov = acc3[r] + b_f[3*KHH + j];
        float c_old = cst[m*KHH + j];
        float c_new = sigmoidf_(fv)*c_old + sigmoidf_(iv)*tanhf(gv);
        float h_new = sigmoidf_(ov)*tanhf(c_new);
        cnewA[jt][r] = c_new;
        hnewA[jt][r] = h_new;
      }
    }
    __syncthreads();
#pragma unroll
    for (int jt = 0; jt < 3; ++jt){
      int n = lane & 15, hi = lane >> 4;
      int j = j0 + jt*16 + n;
#pragma unroll
      for (int r = 0; r < 8; ++r){
        int m = r + 8*hi;
        if (t < meta[16 + m]){
          cst[m*KHH + j] = cnewA[jt][r];
          hbf[m*KHH + j] = (bf16)hnewA[jt][r];
        }
      }
    }
    __syncthreads();
  }

  for (int i = tid; i < 16*KHH; i += blockDim.x){
    int m = i / KHH, j = i - m*KHH;
    out[(size_t)(base + m)*KH + j] = (float)hbf[i];
  }
  for (int i = tid; i < 16*KH; i += blockDim.x){
    int m = i / KH, k = i - m*KH;
    int row = min(meta[m] + meta[16 + m] - 1, KT - 1);
    xs[i] = (bf16)hs[((size_t)meta[32 + m]*KT + row)*KH + k];
  }
  __syncthreads();

#pragma unroll
  for (int jt = 0; jt < 3; ++jt){
    int jc = j0 + jt*16;
    v8f a0 = {}, a2 = {}, a3 = {};
    for (int kk = 0; kk < KH; kk += 32){
      v16bf a = load_a_frag(xs + kk, KH, lane);
      a0 = wmma_bf16(a, load_bT_frag(wih_b + (size_t)(jc)*KH       + kk, KH, lane), a0);
      a2 = wmma_bf16(a, load_bT_frag(wih_b + (size_t)(2*KHH+jc)*KH + kk, KH, lane), a2);
      a3 = wmma_bf16(a, load_bT_frag(wih_b + (size_t)(3*KHH+jc)*KH + kk, KH, lane), a3);
    }
    int n = lane & 15, hi = lane >> 4;
    int j = jc + n;
#pragma unroll
    for (int r = 0; r < 8; ++r){
      int m = r + 8*hi;
      float iv = a0[r] + b_b[j];
      float gv = a2[r] + b_b[2*KHH + j];
      float ov = a3[r] + b_b[3*KHH + j];
      float cb = sigmoidf_(iv)*tanhf(gv);
      out[(size_t)(base + m)*KH + KHH + j] = sigmoidf_(ov)*tanhf(cb);
    }
  }
}

// ---------------------------------------------------------------------------
// Cross attention + residual + LayerNorm, one block per batch (S=24, pad 32).
// ---------------------------------------------------------------------------
__global__ void cross_attn_ln_kernel(float* __restrict__ qio, const float* __restrict__ kv,
                                     const bf16* __restrict__ in_w, const float* __restrict__ in_b,
                                     const bf16* __restrict__ out_w, const float* __restrict__ out_b,
                                     const float* __restrict__ ln_g, const float* __restrict__ ln_b){
  extern __shared__ char smem[];
  bf16*  qa  = (bf16*)smem;                 // [32][768]  q input; reused for O
  bf16*  kb  = qa + 32*KH;                  // [32][768]
  bf16*  qkv = kb + 32*KH;                  // [32][2304]; reused for Y f32
  float* sc  = (float*)(qkv + 32*3*KH);     // [8][32][32]
  bf16*  pb  = (bf16*)(sc + 8*32*32);       // [8][32][32]

  const int b    = blockIdx.x;
  const int tid  = threadIdx.x, wave = tid >> 5, lane = tid & 31;
  const float* qrow = qio + (size_t)b*KS*KH;
  const float* krow = kv  + (size_t)b*KS*KH;

  for (int i = tid; i < 32*KH; i += blockDim.x){
    int m = i / KH, k = i - m*KH;
    qa[i] = (m < KS) ? (bf16)qrow[m*KH + k] : (bf16)0.0f;
    kb[i] = (m < KS) ? (bf16)krow[m*KH + k] : (bf16)0.0f;
  }
  __syncthreads();

  for (int flat = wave; flat < 2*144; flat += 8){
    int m0 = (flat / 144) * 16, n0 = (flat % 144) * 16;
    const bf16* Amat = (n0 < KH) ? qa : kb;
    __builtin_prefetch(in_w + (size_t)n0*KH, 0, 1);
    v8f acc = {};
    for (int kk = 0; kk < KH; kk += 32){
      v16bf a  = load_a_frag(Amat + (size_t)m0*KH + kk, KH, lane);
      v16bf bb = load_bT_frag(in_w + (size_t)n0*KH + kk, KH, lane);
      acc = wmma_bf16(a, bb, acc);
    }
    int n = lane & 15, hi = lane >> 4;
#pragma unroll
    for (int r = 0; r < 8; ++r){
      int m = r + 8*hi;
      qkv[(size_t)(m0 + m)*(3*KH) + n0 + n] = (bf16)(acc[r] + in_b[n0 + n]);
    }
  }
  __syncthreads();

  const float scale = 0.1020620726159658f;     // 1/sqrt(96)
  {
    int h = wave;
#pragma unroll
    for (int t4 = 0; t4 < 4; ++t4){
      int m0 = (t4 >> 1)*16, n0 = (t4 & 1)*16;
      v8f acc = {};
      for (int kk = 0; kk < KHD; kk += 32){
        v16bf a  = load_a_frag (qkv + (size_t)m0*(3*KH) + h*KHD + kk, 3*KH, lane);
        v16bf bb = load_bT_frag(qkv + (size_t)n0*(3*KH) + KH + h*KHD + kk, 3*KH, lane);
        acc = wmma_bf16(a, bb, acc);
      }
      int n = lane & 15, hi = lane >> 4;
#pragma unroll
      for (int r = 0; r < 8; ++r){
        int m   = r + 8*hi;
        int col = n0 + n;
        sc[h*1024 + (m0 + m)*32 + col] = (col < KS) ? acc[r]*scale : -1e30f;
      }
    }
  }
  __syncthreads();

  {
    int h = tid >> 5, row = tid & 31;
    float mx = -1e30f;
    for (int c = 0; c < 32; ++c) mx = fmaxf(mx, sc[h*1024 + row*32 + c]);
    float e[32], sum = 0.f;
    for (int c = 0; c < 32; ++c){ e[c] = __expf(sc[h*1024 + row*32 + c] - mx); sum += e[c]; }
    float inv = 1.0f / sum;
    for (int c = 0; c < 32; ++c) pb[h*1024 + row*32 + c] = (bf16)(e[c]*inv);
  }
  __syncthreads();

  {
    int h = wave;
#pragma unroll
    for (int t6 = 0; t6 < 12; ++t6){
      int m0 = (t6 / 6)*16, n0 = (t6 % 6)*16;
      v8f acc = {};
      v16bf a  = load_a_frag(pb + h*1024 + m0*32, 32, lane);
      v16bf bb = load_b_frag(qkv + 2*KH + h*KHD + n0, 3*KH, lane);
      acc = wmma_bf16(a, bb, acc);
      int n = lane & 15, hi = lane >> 4;
#pragma unroll
      for (int r = 0; r < 8; ++r){
        int m = r + 8*hi;
        qa[(size_t)(m0 + m)*KH + h*KHD + n0 + n] = (bf16)acc[r];
      }
    }
  }
  __syncthreads();

  float* yf = (float*)qkv;
  for (int flat = wave; flat < 2*48; flat += 8){
    int m0 = (flat / 48)*16, n0 = (flat % 48)*16;
    v8f acc = {};
    for (int kk = 0; kk < KH; kk += 32){
      v16bf a  = load_a_frag (qa + (size_t)m0*KH + kk, KH, lane);
      v16bf bb = load_bT_frag(out_w + (size_t)n0*KH + kk, KH, lane);
      acc = wmma_bf16(a, bb, acc);
    }
    int n = lane & 15, hi = lane >> 4;
#pragma unroll
    for (int r = 0; r < 8; ++r){
      int m = r + 8*hi;
      yf[(m0 + m)*KH + n0 + n] = acc[r] + out_b[n0 + n];
    }
  }
  __syncthreads();

  for (int row = wave; row < KS; row += 8){
    float sum = 0.f, sq = 0.f;
    for (int k = lane; k < KH; k += 32){
      float v = yf[row*KH + k] + qrow[row*KH + k];
      sum += v; sq += v*v;
    }
#pragma unroll
    for (int off = 16; off > 0; off >>= 1){
      sum += __shfl_xor(sum, off, 32);
      sq  += __shfl_xor(sq,  off, 32);
    }
    float mu  = sum * (1.0f/KH);
    float var = sq  * (1.0f/KH) - mu*mu;
    float rs  = rsqrtf(var + 1e-5f);
    for (int k = lane; k < KH; k += 32){
      float v = yf[row*KH + k] + qrow[row*KH + k];
      qio[(size_t)b*KS*KH + row*KH + k] = (v - mu)*rs*ln_g[k] + ln_b[k];
    }
  }
}

// ---------------------------------------------------------------------------
// apply_spans: one block per (batch, span); only covered tokens run the MLP.
// ---------------------------------------------------------------------------
__global__ void span_apply_kernel(float* __restrict__ tok, const int* __restrict__ spans,
                                  const float* __restrict__ reps,
                                  const bf16* __restrict__ w1, const float* __restrict__ b1,
                                  const bf16* __restrict__ w2, const float* __restrict__ b2){
  extern __shared__ char smem[];
  bf16* A  = (bf16*)smem;          // [16][1536]
  bf16* h1 = A + 16*2*KH;          // [16][768]

  const int g = blockIdx.x;
  const int b = g / KS, s = g % KS;
  int st = max(spans[(b*KS + s)*2 + 0], 0);
  int en = min(spans[(b*KS + s)*2 + 1], KT);
  int w  = en - st;
  if (w <= 0) return;
  w = min(w, KW);

  const int tid = threadIdx.x, wave = tid >> 5, lane = tid & 31;

  for (int i = tid; i < 16*2*KH; i += blockDim.x){
    int m = i / (2*KH), c = i - m*(2*KH);
    float v = 0.f;
    if (m < w)
      v = (c < KH) ? tok[((size_t)b*KT + st + m)*KH + c]
                   : reps[((size_t)b*KS + s)*KH + (c - KH)];
    A[i] = (bf16)v;
  }
  __syncthreads();

  for (int nt = wave; nt < 48; nt += 8){
    int n0 = nt*16;
    __builtin_prefetch(w1 + n0, 0, 1);
    v8f acc = {};
    for (int kk = 0; kk < 2*KH; kk += 32){
      v16bf a  = load_a_frag(A + kk, 2*KH, lane);
      v16bf bb = load_b_frag(w1 + (size_t)kk*KH + n0, KH, lane);
      acc = wmma_bf16(a, bb, acc);
    }
    int n = lane & 15, hi = lane >> 4;
#pragma unroll
    for (int r = 0; r < 8; ++r){
      int m = r + 8*hi;
      h1[m*KH + n0 + n] = (bf16)fmaxf(acc[r] + b1[n0 + n], 0.f);
    }
  }
  __syncthreads();

  for (int nt = wave; nt < 48; nt += 8){
    int n0 = nt*16;
    v8f acc = {};
    for (int kk = 0; kk < KH; kk += 32){
      v16bf a  = load_a_frag(h1 + kk, KH, lane);
      v16bf bb = load_b_frag(w2 + (size_t)kk*KH + n0, KH, lane);
      acc = wmma_bf16(a, bb, acc);
    }
    int n = lane & 15, hi = lane >> 4;
#pragma unroll
    for (int r = 0; r < 8; ++r){
      int m = r + 8*hi;
      if (m < w)
        tok[((size_t)b*KT + st + m)*KH + n0 + n] = acc[r] + b2[n0 + n];
    }
  }
}

// ---------------------------------------------------------------------------
// host launcher
// ---------------------------------------------------------------------------
extern "C" void kernel_launch(void* const* d_in, const int* in_sizes, int n_in,
                              void* d_out, int out_size, void* d_ws, size_t ws_size,
                              hipStream_t stream){
  (void)in_sizes; (void)n_in; (void)out_size;

  const float* hs        = (const float*)d_in[0];
  const int*   a_spans   = (const int*)  d_in[1];
  const int*   o_spans   = (const int*)  d_in[2];
  const float* a_wih_f   = (const float*)d_in[3];
  const float* a_whh_f   = (const float*)d_in[4];
  const float* a_b_f     = (const float*)d_in[5];
  const float* a_wih_b   = (const float*)d_in[6];
  const float* a_b_b     = (const float*)d_in[8];
  const float* o_wih_f   = (const float*)d_in[9];
  const float* o_whh_f   = (const float*)d_in[10];
  const float* o_b_f     = (const float*)d_in[11];
  const float* o_wih_b   = (const float*)d_in[12];
  const float* o_b_b     = (const float*)d_in[14];
  const float* ao_in_w   = (const float*)d_in[15];
  const float* ao_in_b   = (const float*)d_in[16];
  const float* ao_out_w  = (const float*)d_in[17];
  const float* ao_out_b  = (const float*)d_in[18];
  const float* oa_in_w   = (const float*)d_in[19];
  const float* oa_in_b   = (const float*)d_in[20];
  const float* oa_out_w  = (const float*)d_in[21];
  const float* oa_out_b  = (const float*)d_in[22];
  const float* ln_a_g    = (const float*)d_in[23];
  const float* ln_a_b    = (const float*)d_in[24];
  const float* ln_o_g    = (const float*)d_in[25];
  const float* ln_o_b    = (const float*)d_in[26];
  const float* p_w1      = (const float*)d_in[27];
  const float* p_b1      = (const float*)d_in[28];
  const float* p_w2      = (const float*)d_in[29];
  const float* p_b2      = (const float*)d_in[30];
  float* out = (float*)d_out;

  char* ws = (char*)d_ws;
  size_t off = 0;
  auto carveB = [&](size_t elems)->bf16*{
    bf16* p = (bf16*)(ws + off);
    off += ((elems*sizeof(bf16) + 255) / 256) * 256;
    return p;
  };
  auto carveF = [&](size_t elems)->float*{
    float* p = (float*)(ws + off);
    off += ((elems*sizeof(float) + 255) / 256) * 256;
    return p;
  };

  bf16* a_wih_f_b  = carveB((size_t)1536*768);
  bf16* a_whh_f_b  = carveB((size_t)1536*384);
  bf16* a_wih_b_b  = carveB((size_t)1536*768);
  bf16* o_wih_f_b  = carveB((size_t)1536*768);
  bf16* o_whh_f_b  = carveB((size_t)1536*384);
  bf16* o_wih_b_b  = carveB((size_t)1536*768);
  bf16* ao_in_w_b  = carveB((size_t)2*2304*768);
  bf16* ao_out_w_b = carveB((size_t)2*768*768);
  bf16* oa_in_w_b  = carveB((size_t)2*2304*768);
  bf16* oa_out_w_b = carveB((size_t)2*768*768);
  bf16* p_w1_b     = carveB((size_t)1536*768);
  bf16* p_w2_b     = carveB((size_t)768*768);
  float* ea        = carveF((size_t)KNS*KH);
  float* eo        = carveF((size_t)KNS*KH);

  // optional precompute-path buffers (carved last so fallback needs less ws)
  bf16*  Xg_a = carveB((size_t)KNS*KW*KH);
  bf16*  Xg_o = carveB((size_t)KNS*KW*KH);
  bf16*  Xb_a = carveB((size_t)KNS*KH);
  bf16*  Xb_o = carveB((size_t)KNS*KH);
  float* Gx_a = carveF((size_t)KNS*KW*4*KHH);
  float* Gx_o = carveF((size_t)KNS*KW*4*KHH);
  float* Gb_a = carveF((size_t)KNS*4*KHH);
  float* Gb_o = carveF((size_t)KNS*4*KHH);
  const bool fast_lstm = (off <= ws_size);

  auto cvt = [&](const float* s, bf16* d, int n){
    cvt_f32_to_bf16<<<(n + 255)/256, 256, 0, stream>>>(s, d, n);
  };
  cvt(a_wih_f, a_wih_f_b, 1536*768);
  cvt(a_whh_f, a_whh_f_b, 1536*384);
  cvt(a_wih_b, a_wih_b_b, 1536*768);
  cvt(o_wih_f, o_wih_f_b, 1536*768);
  cvt(o_whh_f, o_whh_f_b, 1536*384);
  cvt(o_wih_b, o_wih_b_b, 1536*768);
  cvt(ao_in_w,  ao_in_w_b,  2*2304*768);
  cvt(ao_out_w, ao_out_w_b, 2*768*768);
  cvt(oa_in_w,  oa_in_w_b,  2*2304*768);
  cvt(oa_out_w, oa_out_w_b, 2*768*768);
  cvt(p_w1, p_w1_b, 1536*768);
  cvt(p_w2, p_w2_b, 768*768);

  if (fast_lstm){
    // 1) gather span windows (both encoders)
    lstm_gather_kernel<<<KNS, 256, 0, stream>>>(hs, a_spans, Xg_a, Xb_a);
    lstm_gather_kernel<<<KNS, 256, 0, stream>>>(hs, o_spans, Xg_o, Xb_o);
    // 2) hoisted input projections: Gx = Xg @ Wih^T  (M=6144, N=1536, K=768)
    const size_t gsmem = (size_t)2*128*GKT*sizeof(bf16);
    dim3 gx_grid(1536/64, (KNS*KW)/128);
    gemm_a_wt_kernel<<<gx_grid, 256, gsmem, stream>>>(Xg_a, a_wih_f_b, Gx_a, KNS*KW, 1536, KH);
    gemm_a_wt_kernel<<<gx_grid, 256, gsmem, stream>>>(Xg_o, o_wih_f_b, Gx_o, KNS*KW, 1536, KH);
    //    backward-cell projections: Gb = x_last @ Wih_b^T (M=384)
    dim3 gb_grid(1536/64, KNS/128);
    gemm_a_wt_kernel<<<gb_grid, 256, gsmem, stream>>>(Xb_a, a_wih_b_b, Gb_a, KNS, 1536, KH);
    gemm_a_wt_kernel<<<gb_grid, 256, gsmem, stream>>>(Xb_o, o_wih_b_b, Gb_o, KNS, 1536, KH);
    // 3) short recurrence, both encoders in one launch
    const size_t rsmem = (size_t)16*KHH*sizeof(bf16) + (size_t)16*KHH*sizeof(float) + 16*sizeof(int);
    lstm_rec_kernel<<<48, 256, rsmem, stream>>>(a_spans, o_spans, Gx_a, Gx_o,
                                                a_whh_f_b, o_whh_f_b, a_b_f, o_b_f, ea, eo);
    // 4) backward cells (elementwise)
    const int nbw = (KNS*KHH + 255)/256;
    lstm_bwd_kernel<<<nbw, 256, 0, stream>>>(Gb_a, a_b_b, ea);
    lstm_bwd_kernel<<<nbw, 256, 0, stream>>>(Gb_o, o_b_b, eo);
  } else {
    const size_t lstm_smem = (size_t)16*KH*sizeof(bf16) + (size_t)16*KHH*sizeof(bf16)
                           + (size_t)16*KHH*sizeof(float) + 48*sizeof(int);
    lstm_span_kernel<<<KNS/16, 256, lstm_smem, stream>>>(
        hs, a_spans, a_wih_f_b, a_whh_f_b, a_b_f, a_wih_b_b, a_b_b, ea);
    lstm_span_kernel<<<KNS/16, 256, lstm_smem, stream>>>(
        hs, o_spans, o_wih_f_b, o_whh_f_b, o_b_f, o_wih_b_b, o_b_b, eo);
  }

  // ---- L=2 rounds of cross attention + LN ----
  const size_t attn_smem = (size_t)32*KH*sizeof(bf16)*2 + (size_t)32*3*KH*sizeof(bf16)
                         + (size_t)8*32*32*sizeof(float) + (size_t)8*32*32*sizeof(bf16);
  for (int l = 0; l < 2; ++l){
    cross_attn_ln_kernel<<<KB, 256, attn_smem, stream>>>(
        ea, eo,
        ao_in_w_b  + (size_t)l*2304*768, ao_in_b  + (size_t)l*2304,
        ao_out_w_b + (size_t)l*768*768,  ao_out_b + (size_t)l*768,
        ln_a_g + (size_t)l*768, ln_a_b + (size_t)l*768);
    cross_attn_ln_kernel<<<KB, 256, attn_smem, stream>>>(
        eo, ea,
        oa_in_w_b  + (size_t)l*2304*768, oa_in_b  + (size_t)l*2304,
        oa_out_w_b + (size_t)l*768*768,  oa_out_b + (size_t)l*768,
        ln_o_g + (size_t)l*768, ln_o_b + (size_t)l*768);
  }

  // ---- bulk copy hidden -> out ----
  const int n4 = (KB*KT*KH) / 4;
  copy_f32x4<<<(n4 + 255)/256, 256, 0, stream>>>((const float4*)hs, (float4*)out, n4);

  // ---- apply spans (aspect first, then opinion on updated tokens) ----
  const size_t span_smem = (size_t)16*2*KH*sizeof(bf16) + (size_t)16*KH*sizeof(bf16);
  span_apply_kernel<<<KB*KS, 256, span_smem, stream>>>(out, a_spans, ea, p_w1_b, p_b1, p_w2_b, p_b2);
  span_apply_kernel<<<KB*KS, 256, span_smem, stream>>>(out, o_spans, eo, p_w1_b, p_b1, p_w2_b, p_b2);
}
